// TextEncoderBlock_67310727463215
// MI455X (gfx1250) — compile-verified, matrix operand order checked by reading, not run on hardware
//
#include <hip/hip_runtime.h>
#include <hip/hip_bf16.h>

#define EMBD   960
#define FFN    2560
#define HD     64
#define KVH    5
#define QH     15
#define LSEQ   2048
#define KVD    (KVH*HD)   // 320
#define QD     (QH*HD)    // 960

typedef __attribute__((ext_vector_type(16))) __bf16 bf16x16;
typedef __attribute__((ext_vector_type(8)))  __bf16 bf16x8;
typedef __attribute__((ext_vector_type(8)))  float  f32x8;

__device__ __forceinline__ f32x8 wmma_bf16(bf16x16 a, bf16x16 b, f32x8 c) {
    return __builtin_amdgcn_wmma_f32_16x16x32_bf16(
        false, a, false, b, (short)0, c, false, false);
}

// 16-bit A-style fragment from a row-major matrix (also used as B via the
// transposed-operand trick). lanes 0-15: elems 0..7 -> K=k0+0..7,
// 8..15 -> K=k0+16..23 ; lanes 16-31: +8 on both ranges.
__device__ __forceinline__ bf16x16 load_frag(const __bf16* base, int row,
                                             int stride, int k0, int lane) {
    const __bf16* p = base + (size_t)row * stride + k0 + ((lane & 16) ? 8 : 0);
    bf16x8 lo = *(const bf16x8*)(p);
    bf16x8 hi = *(const bf16x8*)(p + 16);
    return __builtin_shufflevector(lo, hi, 0,1,2,3,4,5,6,7,8,9,10,11,12,13,14,15);
}

// Same fragment pattern but from an LDS row of 32 bf16 (row stride fixed = 32).
__device__ __forceinline__ bf16x16 lds_frag(const __bf16* rowbase, int koff) {
    bf16x8 lo = *(const bf16x8*)(rowbase + koff);
    bf16x8 hi = *(const bf16x8*)(rowbase + koff + 16);
    return __builtin_shufflevector(lo, hi, 0,1,2,3,4,5,6,7,8,9,10,11,12,13,14,15);
}

// Low 32 bits of a generic pointer to a __shared__ object == LDS byte offset
// (AS3 -> generic addrspacecast keeps the offset in the low word).
__device__ __forceinline__ uint32_t lds_off_u32(const void* p) {
    return (uint32_t)(uintptr_t)p;
}

// ---------------------------------------------------------------------------
// B-tile staging into LDS: TDM (tensor_load_to_lds) when the builtin exists,
// else per-lane async global->LDS copies. Tile is 64 rows x 32 bf16 (4 KB).
// ---------------------------------------------------------------------------
#if __has_builtin(__builtin_amdgcn_tensor_load_to_lds)

typedef unsigned int u32x4 __attribute__((ext_vector_type(4)));
typedef int          i32x4 __attribute__((ext_vector_type(4)));
typedef int          i32x8 __attribute__((ext_vector_type(8)));

__device__ __forceinline__ void tdm_load_tile(uint32_t lds_bytes,
                                              const void* gaddr,
                                              uint32_t K_elems, uint32_t N_rows) {
    uint64_t ga = (uint64_t)(uintptr_t)gaddr;
    u32x4 g0;
    g0[0] = 1u;                                   // count=1 (valid user D#)
    g0[1] = lds_bytes;                            // lds_addr
    g0[2] = (uint32_t)ga;                         // global_addr[31:0]
    g0[3] = (uint32_t)((ga >> 32) & 0x01FFFFFFu)  // global_addr[56:32]
          | (2u << 30);                           // type=2 ("image")
    i32x8 g1;
    g1[0] = 0x00010000;                           // data_size=2B; no pad/iter
    g1[1] = (int)((K_elems & 0xFFFFu) << 16);     // tensor_dim0[15:0]
    g1[2] = (int)((K_elems >> 16) | ((N_rows & 0xFFFFu) << 16));
    g1[3] = (int)((N_rows >> 16) | (32u << 16));  // tile_dim0 = 32
    g1[4] = 64;                                   // tile_dim1 = 64, tile_dim2=0
    g1[5] = (int)K_elems;                         // tensor_dim0_stride[31:0]
    g1[6] = 0;                                    // stride0 hi, stride1 lo
    g1[7] = 0;
    i32x4 gz = {0, 0, 0, 0};
#if __clang_major__ >= 23
    i32x8 gz8 = {0, 0, 0, 0, 0, 0, 0, 0};
    __builtin_amdgcn_tensor_load_to_lds(g0, g1, gz, gz, gz8, 0);
#else
    __builtin_amdgcn_tensor_load_to_lds(g0, g1, gz, gz, 0);
#endif
}

__device__ __forceinline__ void stage_tile(__bf16* dstLds, const __bf16* Bt,
                                           int n0, int kt, int K, int N,
                                           int tid, int wave) {
    if (wave == 0)
        tdm_load_tile(lds_off_u32(dstLds), Bt + (size_t)n0 * K + kt,
                      (uint32_t)K, (uint32_t)N);
    (void)tid;
}
__device__ __forceinline__ void stage_fence(int wave) {
    if (wave == 0) __builtin_amdgcn_s_wait_tensorcnt(0);
    __syncthreads();
}

#else  // fallback: per-lane async copies (one b128 per lane, 4 KB per block)

__device__ __forceinline__ void async_copy_b128(uint32_t lds_bytes, uint64_t g) {
    asm volatile("global_load_async_to_lds_b128 %0, %1, off"
                 :: "v"(lds_bytes), "v"(g) : "memory");
}
__device__ __forceinline__ void stage_tile(__bf16* dstLds, const __bf16* Bt,
                                           int n0, int kt, int K, int N,
                                           int tid, int wave) {
    const int srow = tid >> 2;          // 0..63
    const int sseg = (tid & 3) * 8;     // 0,8,16,24 elements
    async_copy_b128(lds_off_u32(dstLds + srow * 32 + sseg),
                    (uint64_t)(uintptr_t)(Bt + (size_t)(n0 + srow) * K + kt + sseg));
    (void)N; (void)wave;
}
__device__ __forceinline__ void stage_fence(int) {
    asm volatile("s_wait_asynccnt 0x0" ::: "memory");
    __syncthreads();
}

#endif

// ---------------------------------------------------------------------------
// GEMM: C[M,N] (f32, += residual) = A_bf16[M,K] x (Bt_bf16[N,K])^T
// wave tile 32x64 (2x4 accs, 8 WMMAs per k-step), 8 waves -> 256x64 block
// tile. B-tile (64x32) double-buffered in LDS via TDM / async copies.
// M % 256 == 0, N % 64 == 0, K % 32 == 0.
// ---------------------------------------------------------------------------
__global__ void gemm_bf16_wmma(const __bf16* __restrict__ A,
                               const __bf16* __restrict__ Bt,
                               float* __restrict__ C,
                               const float* __restrict__ residual,
                               int M, int N, int K) {
    __shared__ __align__(16) __bf16 Bs[2][64 * 32];
    const int tid  = threadIdx.x;
    const int lane = tid & 31;
    const int wave = tid >> 5;
    const int m0 = blockIdx.x * 256 + wave * 32;
    const int n0 = blockIdx.y * 64;
    const int l15  = lane & 15;
    const int koff = (lane & 16) ? 8 : 0;

    stage_tile(Bs[0], Bt, n0, 0, K, N, tid, wave);
    stage_fence(wave);

    f32x8 acc[2][4] = {};
    bf16x16 a0 = load_frag(A, m0 + l15,      K, 0, lane);
    bf16x16 a1 = load_frag(A, m0 + 16 + l15, K, 0, lane);

    int cur = 0;
    for (int kt = 0; kt < K; kt += 32) {
        const bool has_next = (kt + 32) < K;
        if (has_next)
            stage_tile(Bs[cur ^ 1], Bt, n0, kt + 32, K, N, tid, wave);

        bf16x16 a0n, a1n;
        if (has_next) {
            if (kt + 64 < K)
                __builtin_prefetch(A + (size_t)(m0 + l15) * K + kt + 64, 0, 1);
            a0n = load_frag(A, m0 + l15,      K, kt + 32, lane);
            a1n = load_frag(A, m0 + 16 + l15, K, kt + 32, lane);
        }

        #pragma unroll
        for (int j = 0; j < 4; ++j) {
            bf16x16 b = lds_frag(&Bs[cur][(j * 16 + l15) * 32], koff);
            acc[0][j] = wmma_bf16(a0, b, acc[0][j]);
            acc[1][j] = wmma_bf16(a1, b, acc[1][j]);
        }

        if (has_next) { a0 = a0n; a1 = a1n; stage_fence(wave); cur ^= 1; }
    }

    const int rbase = (lane & 16) ? 8 : 0;
    #pragma unroll
    for (int i = 0; i < 2; ++i) {
        #pragma unroll
        for (int j = 0; j < 4; ++j) {
            #pragma unroll
            for (int r = 0; r < 8; ++r) {
                size_t idx = (size_t)(m0 + i * 16 + rbase + r) * N
                           + (n0 + j * 16 + l15);
                float v = acc[i][j][r];
                if (residual) v += residual[idx];
                C[idx] = v;
            }
        }
    }
}

// ---------------------------------------------------------------------------
// Flash attention, transposed-scores form (see round-1 notes): S^T = K x Q^T,
// softmax over keys lives in accumulator rows; the C-layout of S^T equals the
// B-fragment layout of P^T, so ctx^T = V^T x P^T needs zero lane shuffles.
// ---------------------------------------------------------------------------
__global__ void attn_wmma(const __bf16* __restrict__ qb,   // [L, QH*64] roped
                          const __bf16* __restrict__ kb,   // [L, KVH*64] roped
                          const __bf16* __restrict__ vT,   // [KVH][64][L]
                          __bf16* __restrict__ ctx) {      // [L, QH*64]
    const int lane = threadIdx.x;
    const int qt = blockIdx.x;
    const int h  = blockIdx.y;
    const int hk = (h * KVH) / QH;
    const int q0 = qt * 16;
    const int l15 = lane & 15;
    const __bf16* vTh = vT + (size_t)hk * HD * LSEQ;

    bf16x16 qf0 = load_frag(qb, q0 + l15, QD, h * HD + 0,  lane);
    bf16x16 qf1 = load_frag(qb, q0 + l15, QD, h * HD + 32, lane);

    f32x8 ctxT0 = {}, ctxT1 = {}, ctxT2 = {}, ctxT3 = {};
    float m_run = -1e30f, l_run = 0.f;
    const float scale = 0.125f;
    const int q = q0 + l15;
    const int kend = q0 + 16;

    for (int kt = 0; kt < kend; kt += 32) {
        f32x8 s0 = {}, s1 = {};
        {
            bf16x16 ka = load_frag(kb, kt + l15, KVD, hk * HD + 0, lane);
            s0 = wmma_bf16(ka, qf0, s0);
            ka = load_frag(kb, kt + l15, KVD, hk * HD + 32, lane);
            s0 = wmma_bf16(ka, qf1, s0);
            ka = load_frag(kb, kt + 16 + l15, KVD, hk * HD + 0, lane);
            s1 = wmma_bf16(ka, qf0, s1);
            ka = load_frag(kb, kt + 16 + l15, KVD, hk * HD + 32, lane);
            s1 = wmma_bf16(ka, qf1, s1);
        }
        const int kb0 = kt + ((lane & 16) ? 8 : 0);
        const int kb1 = kb0 + 16;
        float tm = -1e30f;
        #pragma unroll
        for (int r = 0; r < 8; ++r) {
            float v0 = s0[r] * scale; if (kb0 + r > q) v0 = -1e30f;
            float v1 = s1[r] * scale; if (kb1 + r > q) v1 = -1e30f;
            s0[r] = v0; s1[r] = v1;
            tm = fmaxf(tm, fmaxf(v0, v1));
        }
        tm = fmaxf(tm, __shfl_xor(tm, 16, 32));
        const float m_new = fmaxf(m_run, tm);
        const float corr  = __expf(m_run - m_new);
        float lsum = 0.f;
        bf16x16 pfrag;
        #pragma unroll
        for (int r = 0; r < 8; ++r) {
            float p0 = __expf(s0[r] - m_new);
            float p1 = __expf(s1[r] - m_new);
            lsum += p0 + p1;
            pfrag[r]     = (__bf16)p0;
            pfrag[r + 8] = (__bf16)p1;
        }
        lsum += __shfl_xor(lsum, 16, 32);
        l_run = l_run * corr + lsum;
        m_run = m_new;

        ctxT0 *= corr; ctxT1 *= corr; ctxT2 *= corr; ctxT3 *= corr;
        bf16x16 vf;
        vf = load_frag(vTh,  0 + l15, LSEQ, kt, lane); ctxT0 = wmma_bf16(vf, pfrag, ctxT0);
        vf = load_frag(vTh, 16 + l15, LSEQ, kt, lane); ctxT1 = wmma_bf16(vf, pfrag, ctxT1);
        vf = load_frag(vTh, 32 + l15, LSEQ, kt, lane); ctxT2 = wmma_bf16(vf, pfrag, ctxT2);
        vf = load_frag(vTh, 48 + l15, LSEQ, kt, lane); ctxT3 = wmma_bf16(vf, pfrag, ctxT3);
    }

    const float inv_l = 1.f / l_run;
    const int rb = (lane & 16) ? 8 : 0;
    __bf16* dst = ctx + (size_t)q * QD + h * HD;
    f32x8 cs[4] = {ctxT0, ctxT1, ctxT2, ctxT3};
    #pragma unroll
    for (int dt = 0; dt < 4; ++dt)
        #pragma unroll
        for (int r = 0; r < 8; ++r)
            dst[dt * 16 + rb + r] = (__bf16)(cs[dt][r] * inv_l);
}

// ---------------------------------------------------------------------------
// Helpers
// ---------------------------------------------------------------------------
__global__ void rmsnorm_bf16_kernel(const float* __restrict__ x,
                                    const float* __restrict__ g,
                                    __bf16* __restrict__ out, int D) {
    const int row = blockIdx.x;
    const float* xr = x + (size_t)row * D;
    float s = 0.f;
    for (int i = threadIdx.x; i < D; i += blockDim.x) { float v = xr[i]; s += v * v; }
    for (int off = 16; off > 0; off >>= 1) s += __shfl_down(s, off, 32);
    __shared__ float red[8];
    const int lane = threadIdx.x & 31, w = threadIdx.x >> 5;
    if (lane == 0) red[w] = s;
    __syncthreads();
    if (w == 0) {
        float t = (lane < 8) ? red[lane] : 0.f;
        for (int off = 4; off > 0; off >>= 1) t += __shfl_down(t, off, 32);
        if (lane == 0) red[0] = rsqrtf(t / (float)D + 1.1920929e-07f);
    }
    __syncthreads();
    const float r = red[0];
    for (int i = threadIdx.x; i < D; i += blockDim.x)
        out[(size_t)row * D + i] = (__bf16)(xr[i] * r * g[i]);
}

__global__ void transpose_to_bf16(const float* __restrict__ W,
                                  __bf16* __restrict__ Wt, int K, int N) {
    __shared__ float tile[32][33];
    const int kb = blockIdx.y * 32, nb = blockIdx.x * 32;
    const int tx = threadIdx.x & 31, ty = threadIdx.x >> 5;
    for (int i = ty; i < 32; i += 8) {
        int k = kb + i, n = nb + tx;
        tile[i][tx] = (k < K && n < N) ? W[(size_t)k * N + n] : 0.f;
    }
    __syncthreads();
    for (int i = ty; i < 32; i += 8) {
        int n = nb + i, k = kb + tx;
        if (n < N && k < K) Wt[(size_t)n * K + k] = (__bf16)tile[tx][i];
    }
}

__global__ void rope_bf16_kernel(const float* __restrict__ x,
                                 __bf16* __restrict__ out, int n_heads) {
    const int pos = blockIdx.x, h = blockIdx.y, d = threadIdx.x; // 32 threads
    const float* p = x   + ((size_t)pos * n_heads + h) * HD;
    __bf16*     o = out + ((size_t)pos * n_heads + h) * HD;
    float ts = __powf(10000.f, (float)d * (2.f / 64.f));
    float r = (float)pos / ts;
    float s, c; __sincosf(r, &s, &c);
    float x1 = p[d], x2 = p[d + 32];
    o[d]      = (__bf16)(x1 * c - x2 * s);
    o[d + 32] = (__bf16)(x2 * c + x1 * s);
}

__global__ void v_transpose_kernel(const float* __restrict__ v,
                                   __bf16* __restrict__ vT) {
    int idx = blockIdx.x * 256 + threadIdx.x;
    if (idx >= KVH * HD * LSEQ) return;
    int pos = idx % LSEQ;
    int rem = idx / LSEQ;
    int d = rem % HD, h = rem / HD;
    vT[idx] = (__bf16)v[(size_t)pos * KVD + h * HD + d];
}

__global__ void silu_mul_kernel(const float* __restrict__ gt,
                                const float* __restrict__ up,
                                __bf16* __restrict__ out, int n) {
    int i = blockIdx.x * 256 + threadIdx.x;
    if (i < n) {
        float v = gt[i];
        float s = v / (1.f + __expf(-v));
        out[i] = (__bf16)(s * up[i]);
    }
}

// ---------------------------------------------------------------------------
// Host launch
// ---------------------------------------------------------------------------
extern "C" void kernel_launch(void* const* d_in, const int* in_sizes, int n_in,
                              void* d_out, int out_size, void* d_ws, size_t ws_size,
                              hipStream_t stream) {
    const float* x  = (const float*)d_in[0];
    const float* wq = (const float*)d_in[1];
    const float* wk = (const float*)d_in[2];
    const float* wv = (const float*)d_in[3];
    const float* wo = (const float*)d_in[4];
    const float* wg = (const float*)d_in[5];
    const float* wu = (const float*)d_in[6];
    const float* wd = (const float*)d_in[7];
    const float* g1 = (const float*)d_in[8];
    const float* g2 = (const float*)d_in[9];

    float* out_f = (float*)d_out;
    float* kout  = out_f + (size_t)LSEQ * EMBD;
    float* vout  = kout  + (size_t)LSEQ * KVD;

    char* ws = (char*)d_ws;
    size_t off = 0;
    auto alloc = [&](size_t bytes) -> void* {
        void* p = ws + off;
        off = (off + bytes + 255) & ~(size_t)255;
        return p;
    };

    __bf16* wqT = (__bf16*)alloc((size_t)QD  * EMBD * 2);
    __bf16* wkT = (__bf16*)alloc((size_t)KVD * EMBD * 2);
    __bf16* wvT = (__bf16*)alloc((size_t)KVD * EMBD * 2);
    __bf16* woT = (__bf16*)alloc((size_t)EMBD * QD  * 2);
    __bf16* wgT = (__bf16*)alloc((size_t)FFN * EMBD * 2);
    __bf16* wuT = (__bf16*)alloc((size_t)FFN * EMBD * 2);
    __bf16* wdT = (__bf16*)alloc((size_t)EMBD * FFN * 2);
    __bf16* hb  = (__bf16*)alloc((size_t)LSEQ * EMBD * 2);
    float*  qf  = (float*) alloc((size_t)LSEQ * QD  * 4);
    float*  kf  = (float*) alloc((size_t)LSEQ * KVD * 4);
    float*  vf  = (float*) alloc((size_t)LSEQ * KVD * 4);
    __bf16* qb  = (__bf16*)alloc((size_t)LSEQ * QD  * 2);
    __bf16* kb  = (__bf16*)alloc((size_t)LSEQ * KVD * 2);
    __bf16* vTb = (__bf16*)alloc((size_t)KVH * HD * LSEQ * 2);
    __bf16* ctx = (__bf16*)alloc((size_t)LSEQ * QD * 2);
    float*  x2  = (float*) alloc((size_t)LSEQ * EMBD * 4);
    float*  gte = (float*) alloc((size_t)LSEQ * FFN * 4);
    float*  upe = (float*) alloc((size_t)LSEQ * FFN * 4);
    __bf16* act = (__bf16*)alloc((size_t)LSEQ * FFN * 2);

    auto tpose = [&](const float* W, __bf16* Wt, int K, int N) {
        dim3 g((N + 31) / 32, (K + 31) / 32);
        transpose_to_bf16<<<g, 256, 0, stream>>>(W, Wt, K, N);
    };
    auto gemm = [&](const __bf16* A, const __bf16* Bt, float* C,
                    const float* res, int M, int N, int K) {
        dim3 g(M / 256, N / 64);
        gemm_bf16_wmma<<<g, 256, 0, stream>>>(A, Bt, C, res, M, N, K);
    };

    tpose(wq, wqT, EMBD, QD);
    tpose(wk, wkT, EMBD, KVD);
    tpose(wv, wvT, EMBD, KVD);
    tpose(wo, woT, QD, EMBD);
    tpose(wg, wgT, EMBD, FFN);
    tpose(wu, wuT, EMBD, FFN);
    tpose(wd, wdT, FFN, EMBD);

    rmsnorm_bf16_kernel<<<LSEQ, 256, 0, stream>>>(x, g1, hb, EMBD);

    gemm(hb, wqT, qf, nullptr, LSEQ, QD, EMBD);
    gemm(hb, wkT, kf, nullptr, LSEQ, KVD, EMBD);
    gemm(hb, wvT, vf, nullptr, LSEQ, KVD, EMBD);

    rope_bf16_kernel<<<dim3(LSEQ, QH), 32, 0, stream>>>(qf, qb, QH);
    rope_bf16_kernel<<<dim3(LSEQ, KVH), 32, 0, stream>>>(kf, kb, KVH);
    v_transpose_kernel<<<(KVH * HD * LSEQ + 255) / 256, 256, 0, stream>>>(vf, vTb);

    attn_wmma<<<dim3(LSEQ / 16, QH), 32, 0, stream>>>(qb, kb, vTb, ctx);

    gemm(ctx, woT, x2, x, LSEQ, EMBD, QD);

    rmsnorm_bf16_kernel<<<LSEQ, 256, 0, stream>>>(x2, g2, hb, EMBD);
    gemm(hb, wgT, gte, nullptr, LSEQ, FFN, EMBD);
    gemm(hb, wuT, upe, nullptr, LSEQ, FFN, EMBD);
    silu_mul_kernel<<<(LSEQ * FFN + 255) / 256, 256, 0, stream>>>(gte, upe, act, LSEQ * FFN);
    gemm(act, wdT, out_f, x2, LSEQ, EMBD, FFN);

    rmsnorm_bf16_kernel<<<LSEQ, 256, 0, stream>>>(x2, g1, hb, EMBD);
    gemm(hb, wkT, kout, nullptr, LSEQ, KVD, EMBD);
    gemm(hb, wvT, vout, nullptr, LSEQ, KVD, EMBD);

    (void)in_sizes; (void)n_in; (void)out_size; (void)ws_size;
}
